// NaiveCustomLSTM_25761213841774
// MI455X (gfx1250) — matrix operand order repaired and checked by reading, not compiled, not run on hardware
//
#include <hip/hip_runtime.h>
#include <cstdint>
#include <cmath>

// Problem constants (from reference): S=256, B=32, D=H=1024, 4H=4096, L=2
#define SEQ     256
#define BATCH   32
#define HID     1024
#define FH      4096
#define NLAYERS 2

typedef __attribute__((ext_vector_type(16))) __bf16 v16bf;
typedef __attribute__((ext_vector_type(8)))  __bf16 v8bf;   // 16-byte load unit
typedef __attribute__((ext_vector_type(8)))  float  v8f;
typedef __attribute__((ext_vector_type(4)))  float  v4f;

// ---------------------------------------------------------------- helpers ----
__device__ __forceinline__ __bf16 f2bf(float f) {
  uint32_t u = __builtin_bit_cast(uint32_t, f);
  uint32_t r = u + 0x7FFFu + ((u >> 16) & 1u);   // round-to-nearest-even
  uint16_t h = (uint16_t)(r >> 16);
  return __builtin_bit_cast(__bf16, h);
}

__device__ __forceinline__ float sigmoidf_(float x) {
  return 1.0f / (1.0f + __expf(-x));
}

// A-matrix fragment: 16x32 bf16 (one row per lane, bf16 row-major source).
// ISA layout: lanes 0-15 hold K=0..7,16..23; lanes 16-31 hold K=8..15,24..31.
// kk = 8*(lane>>4). Two aligned 16B loads.
__device__ __forceinline__ v16bf frag_row(const __bf16* __restrict__ p, int kk) {
  v8bf lo = *(const v8bf*)(p + kk);
  v8bf hi = *(const v8bf*)(p + kk + 16);
  return __builtin_shufflevector(lo, hi, 0, 1, 2, 3, 4, 5, 6, 7,
                                         8, 9, 10, 11, 12, 13, 14, 15);
}

// B-matrix fragment: 32x16 bf16 from pre-transposed [N][K] weights.
// ISA layout: lanes 0-15 hold K=0..7,16..23; lanes 16-31 hold K=8..15,24..31.
// koff = 8*(lane>>4). Two aligned 16B loads.
__device__ __forceinline__ v16bf frag_col(const __bf16* __restrict__ p, int koff) {
  v8bf lo = *(const v8bf*)(p + koff);
  v8bf hi = *(const v8bf*)(p + koff + 16);
  return __builtin_shufflevector(lo, hi, 0, 1, 2, 3, 4, 5, 6, 7,
                                         8, 9, 10, 11, 12, 13, 14, 15);
}

__device__ __forceinline__ v8f wmma_bf16(v16bf a, v16bf b, v8f c) {
  return __builtin_amdgcn_wmma_f32_16x16x32_bf16(
      /*neg_a=*/false, a, /*neg_b=*/false, b,
      /*c_mod=*/(short)0, c, /*reuse_a=*/false, /*reuse_b=*/false);
}

// ------------------------------------------------- weight transpose (fp32->bf16)
// W: [K][N] fp32 row-major  ->  WT: [N][K] bf16
__global__ __launch_bounds__(256) void transpose_bf16_kernel(
    const float* __restrict__ W, __bf16* __restrict__ WT, int K, int N) {
  __shared__ __bf16 tile[32][33];
  int kb = blockIdx.y * 32, nb = blockIdx.x * 32;
  int tx = threadIdx.x & 31, ty = threadIdx.x >> 5;   // ty = 0..7
#pragma unroll
  for (int i = 0; i < 32; i += 8)
    tile[ty + i][tx] = f2bf(W[(size_t)(kb + ty + i) * N + nb + tx]);
  __syncthreads();
#pragma unroll
  for (int i = 0; i < 32; i += 8)
    WT[(size_t)(nb + ty + i) * K + kb + tx] = tile[tx][ty + i];
}

// ------------------------------------------------ contiguous fp32 -> bf16 copy
__global__ __launch_bounds__(256) void convert_bf16_kernel(
    const float* __restrict__ src, __bf16* __restrict__ dst, int n8) {
  int i = blockIdx.x * blockDim.x + threadIdx.x;   // one thread = 8 elements
  if (i >= n8) return;
  const v4f* s = (const v4f*)src + 2 * (size_t)i;
  v4f a = s[0], b = s[1];
  v8bf o;
#pragma unroll
  for (int j = 0; j < 4; ++j) { o[j] = f2bf(a[j]); o[j + 4] = f2bf(b[j]); }
  *((v8bf*)dst + (size_t)i) = o;
}

// ---------------------------------------------------------------- state init --
__global__ void init_state_kernel(float* __restrict__ h, float* __restrict__ c,
                                  __bf16* __restrict__ hbf, int n) {
  int i = blockIdx.x * blockDim.x + threadIdx.x;
  if (i < n) { h[i] = 0.0f; c[i] = 0.0f; hbf[i] = __builtin_bit_cast(__bf16, (uint16_t)0); }
}

// ------------------------------------------- xproj = A(bf16) @ WxT(bf16) + bias
// A: [M=8192][1024] bf16 row-major; BT: [FH][1024] bf16; out: [M][FH] fp32.
// Block = 128 threads (4 waves); each wave computes a 32x32 tile (4 WMMAs/k-step).
__global__ __launch_bounds__(128) void gemm_xproj_kernel(
    const __bf16* __restrict__ A, const __bf16* __restrict__ BT,
    const float* __restrict__ bias, float* __restrict__ out) {
  const int wave = threadIdx.x >> 5;
  const int lane = threadIdx.x & 31;
  const int n0 = blockIdx.x * 32;
  const int m0 = (blockIdx.y * 4 + wave) * 32;
  const int half = lane >> 4;
  const int ncol = lane & 15;
  const int kk = half * 8;
  const int koff = half * 8;

  const __bf16* arow0 = A  + (size_t)(m0 + ncol) * HID;
  const __bf16* arow1 = A  + (size_t)(m0 + 16 + ncol) * HID;
  const __bf16* bcol0 = BT + (size_t)(n0 + ncol) * HID;
  const __bf16* bcol1 = BT + (size_t)(n0 + 16 + ncol) * HID;

  v8f acc00 = {}, acc01 = {}, acc10 = {}, acc11 = {};
#pragma unroll 2
  for (int k = 0; k < HID; k += 32) {
    v16bf a0 = frag_row(arow0 + k, kk);
    v16bf a1 = frag_row(arow1 + k, kk);
    v16bf b0 = frag_col(bcol0 + k, koff);
    v16bf b1 = frag_col(bcol1 + k, koff);
    acc00 = wmma_bf16(a0, b0, acc00);
    acc01 = wmma_bf16(a0, b1, acc01);
    acc10 = wmma_bf16(a1, b0, acc10);
    acc11 = wmma_bf16(a1, b1, acc11);
  }
  float bv0 = bias[n0 + ncol];
  float bv1 = bias[n0 + 16 + ncol];
#pragma unroll
  for (int r = 0; r < 8; ++r) {
    int mA = m0 + r + 8 * half;       // C/D layout: lanes 16-31 -> M+8
    int mB = mA + 16;
    out[(size_t)mA * FH + n0 + ncol]      = acc00[r] + bv0;
    out[(size_t)mA * FH + n0 + 16 + ncol] = acc01[r] + bv1;
    out[(size_t)mB * FH + n0 + ncol]      = acc10[r] + bv0;
    out[(size_t)mB * FH + n0 + 16 + ncol] = acc11[r] + bv1;
  }
}

// -------------------------------------------------------- one LSTM time step --
// gates = xproj_t + h_in @ Wh ; block handles a j-tile of 16 columns for ALL 4
// gates (wave w <-> gate w), exchanges via LDS, fuses activations + c/h update.
__global__ __launch_bounds__(128) void lstm_step_kernel(
    const __bf16* __restrict__ hbf_in,   // [32][HID] bf16 (previous h)
    const float* __restrict__ c_in,      // [32][HID]
    float* __restrict__ h_out,           // [32][HID]
    float* __restrict__ c_out,           // [32][HID]
    __bf16* __restrict__ hbf_out,        // [32][HID] bf16 (for next step's GEMM)
    const __bf16* __restrict__ WhT,      // [FH][HID] bf16 (transposed)
    const float* __restrict__ xproj_t,   // [32][FH]
    float* __restrict__ hseq_t,          // [32][HID] fp32 sequence output row t
    __bf16* __restrict__ hseqbf_t) {     // [32][HID] bf16 sequence output row t
  __shared__ float gsm[4][BATCH][16];    // [gate][b][j] pre-activations

  const int wave = threadIdx.x >> 5;     // gate index: 0=f 1=i 2=g 3=o
  const int lane = threadIdx.x & 31;
  const int j0 = blockIdx.x * 16;
  const int gbase = wave * HID + j0;     // column base in 4H gate space
  const int half = lane >> 4;
  const int ncol = lane & 15;
  const int kk = half * 8;
  const int koff = half * 8;

  const __bf16* bcol  = WhT    + (size_t)(gbase + ncol) * HID;
  const __bf16* arow0 = hbf_in + (size_t)ncol * HID;         // rows 0..15
  const __bf16* arow1 = hbf_in + (size_t)(16 + ncol) * HID;  // rows 16..31

  v8f acc0 = {}, acc1 = {};
#pragma unroll 4
  for (int k = 0; k < HID; k += 32) {
    v16bf b  = frag_col(bcol + k, koff);
    v16bf a0 = frag_row(arow0 + k, kk);
    v16bf a1 = frag_row(arow1 + k, kk);
    acc0 = wmma_bf16(a0, b, acc0);
    acc1 = wmma_bf16(a1, b, acc1);
  }

  // add precomputed input projection and stash gate pre-activations in LDS
#pragma unroll
  for (int r = 0; r < 8; ++r) {
    int m0 = r + 8 * half;
    int m1 = 16 + m0;
    gsm[wave][m0][ncol] = acc0[r] + xproj_t[(size_t)m0 * FH + gbase + ncol];
    gsm[wave][m1][ncol] = acc1[r] + xproj_t[(size_t)m1 * FH + gbase + ncol];
  }
  __syncthreads();

  // fused elementwise epilogue: 32*16 = 512 (b,j) cells, 128 threads
  for (int e = threadIdx.x; e < BATCH * 16; e += 128) {
    int m = e >> 4, j = e & 15;
    float f = sigmoidf_(gsm[0][m][j]);
    float i = sigmoidf_(gsm[1][m][j]);
    float g = tanhf(gsm[2][m][j]);
    float o = sigmoidf_(gsm[3][m][j]);
    int col = j0 + j;
    size_t idx = (size_t)m * HID + col;
    float c = f * c_in[idx] + i * g;
    float h = o * tanhf(c);
    __bf16 hb = f2bf(h);
    c_out[idx] = c;
    h_out[idx] = h;
    hbf_out[idx] = hb;
    hseq_t[idx] = h;
    hseqbf_t[idx] = hb;
  }
}

// ------------------------------------------------------------------- driver ----
extern "C" void kernel_launch(void* const* d_in, const int* in_sizes, int n_in,
                              void* d_out, int out_size, void* d_ws, size_t ws_size,
                              hipStream_t stream) {
  (void)in_sizes; (void)n_in; (void)out_size; (void)ws_size;

  const float* x    = (const float*)d_in[0];  // [S][B][D]
  const float* Wx   = (const float*)d_in[1];  // [L][D][FH]
  const float* Wh   = (const float*)d_in[2];  // [L][H][FH]
  const float* bias = (const float*)d_in[3];  // [L][FH]
  float* out = (float*)d_out;  // hidden_seq [S][B][H] ++ new_h [L][B][H] ++ new_c [L][B][H]

  const size_t WSZ = (size_t)FH * HID;        // elements per weight matrix
  const size_t BH  = (size_t)BATCH * HID;
  const size_t SBH = (size_t)SEQ * BH;

  // workspace carve-up (~192.6 MB total)
  char* p = (char*)d_ws;
  __bf16* WxT   = (__bf16*)p;  p += (size_t)NLAYERS * WSZ * sizeof(__bf16);  // 16 MB
  __bf16* WhT   = (__bf16*)p;  p += (size_t)NLAYERS * WSZ * sizeof(__bf16);  // 16 MB
  float*  xproj = (float*)p;   p += SBH * 4 * sizeof(float);                 // 128 MB
  __bf16* xbf   = (__bf16*)p;  p += SBH * sizeof(__bf16);                    // 16 MB
  __bf16* hs1bf = (__bf16*)p;  p += SBH * sizeof(__bf16);                    // 16 MB
  float*  hbuf  = (float*)p;   p += 2 * BH * sizeof(float);                  // ping-pong h
  float*  cbuf  = (float*)p;   p += 2 * BH * sizeof(float);                  // ping-pong c
  __bf16* hbf   = (__bf16*)p;  p += 2 * BH * sizeof(__bf16);                 // ping-pong h (bf16)

  // 1) one-time: transpose+convert weights; convert x to bf16
  dim3 tgrid(FH / 32, HID / 32);
  for (int l = 0; l < NLAYERS; ++l) {
    transpose_bf16_kernel<<<tgrid, 256, 0, stream>>>(Wx + l * WSZ, WxT + l * WSZ, HID, FH);
    transpose_bf16_kernel<<<tgrid, 256, 0, stream>>>(Wh + l * WSZ, WhT + l * WSZ, HID, FH);
  }
  convert_bf16_kernel<<<(int)(SBH / 8 + 255) / 256, 256, 0, stream>>>(x, xbf, (int)(SBH / 8));

  for (int l = 0; l < NLAYERS; ++l) {
    // 2) big parallel GEMM: xproj = X @ Wx_l + b_l   (M=8192, K=1024, N=4096)
    const __bf16* Ain = (l == 0) ? xbf : hs1bf;
    gemm_xproj_kernel<<<dim3(FH / 32, (SEQ * BATCH) / 128), 128, 0, stream>>>(
        Ain, WxT + l * WSZ, bias + (size_t)l * FH, xproj);

    // 3) reset recurrent state
    init_state_kernel<<<(2 * (int)BH + 255) / 256, 256, 0, stream>>>(hbuf, cbuf, hbf, 2 * (int)BH);

    // 4) sequential recurrence: 256 dependent step kernels on the stream.
    //    Layer 0 fp32 seq goes to d_out (fully overwritten by layer 1 later in
    //    stream order); its bf16 seq (hs1bf) feeds layer 1's GEMM.
    __bf16* hseqbf = hs1bf;                 // dead target for l==1 (already consumed)
    for (int t = 0; t < SEQ; ++t) {
      lstm_step_kernel<<<HID / 16, 128, 0, stream>>>(
          hbf + (size_t)(t & 1) * BH, cbuf + (size_t)(t & 1) * BH,
          hbuf + (size_t)((t + 1) & 1) * BH, cbuf + (size_t)((t + 1) & 1) * BH,
          hbf + (size_t)((t + 1) & 1) * BH,
          WhT + l * WSZ,
          xproj + (size_t)t * BATCH * FH,
          out + (size_t)t * BH,
          hseqbf + (size_t)t * BH);
    }

    // 5) final h,c for this layer -> d_out tail sections
    float* hfin = hbuf + (size_t)(SEQ & 1) * BH;
    float* cfin = cbuf + (size_t)(SEQ & 1) * BH;
    hipMemcpyAsync(out + SBH + (size_t)l * BH, hfin,
                   BH * sizeof(float), hipMemcpyDeviceToDevice, stream);
    hipMemcpyAsync(out + SBH + (size_t)NLAYERS * BH + (size_t)l * BH, cfin,
                   BH * sizeof(float), hipMemcpyDeviceToDevice, stream);
  }
}
